// ContextualConv1d_69389491634227
// MI455X (gfx1250) — matrix operand order repaired and checked by reading, not compiled, not run on hardware
//
#include <hip/hip_runtime.h>

typedef __attribute__((ext_vector_type(16))) __bf16 v16bf;
typedef __attribute__((ext_vector_type(8)))  __bf16 v8bf;
typedef __attribute__((ext_vector_type(8)))  float  v8f;
typedef __attribute__((ext_vector_type(4)))  float  v4f;

#define N_      32
#define CIN     256
#define LEN     4096
#define COUT    256
#define KW      9
#define CDIM    128
#define PADW    4
#define CIG     64          // C_in per group
#define COG     64          // C_out per group
#define KKTOT   (KW * CIG)  // 576 = GEMM K per group
#define TILE_L  256         // L strip per block (32 columns per wave, 2 B-tiles)
#define XS_ROWS (TILE_L + KW - 1)  // 264 positions incl. halo (halo=4 each side)
#define XS_P    72                  // Xs pitch in bf16: 144B lane stride -> bank-conflict-free
#define WS_P    584                 // Ws pitch in bf16: 1168B lane stride -> bank-conflict-free

// ---- kernel 1: ctx[n][co] = bias[co] + sum_d c[n][d] * cW[co][d] ----
__global__ __launch_bounds__(256)
void ctx_bias_kernel(const float* __restrict__ c, const float* __restrict__ cw,
                     const float* __restrict__ bias, float* __restrict__ ctx) {
    const int n = blockIdx.x, co = threadIdx.x;
    float s = bias[co];
    const float* cp = c + n * CDIM;
    const float* wp = cw + co * CDIM;
#pragma unroll 8
    for (int d = 0; d < CDIM; ++d) s = fmaf(cp[d], wp[d], s);
    ctx[n * COUT + co] = s;
}

// ---- kernel 2: grouped conv1d as implicit GEMM on bf16 WMMA, f32 accumulate ----
__global__ __launch_bounds__(256)
void conv_wmma_kernel(const float* __restrict__ x, const float* __restrict__ w,
                      const float* __restrict__ ctx, float* __restrict__ out) {
    __shared__ __align__(16) __bf16 Xs[XS_ROWS * XS_P];  // [p][ci], bf16
    __shared__ __align__(16) __bf16 Ws[COG * WS_P];      // [co_r][kk'=k*64+ci], bf16

    const int b    = blockIdx.x;
    const int lb   = b & 15;          // L slab index (4096/256 = 16)
    const int g    = (b >> 4) & 3;    // group
    const int n    = b >> 6;          // batch
    const int tid  = threadIdx.x;
    const int lane = tid & 31;
    const int wv   = tid >> 5;        // wave 0..7 -> 32-column L strip
    const int half = lane >> 4;       // hi/lo 16-lane half
    const int lm   = lane & 15;

    // ---- stage x slab: 16B quads (halo=4 => quads fully valid or fully zero),
    //      f32->bf16, transposed to position-major [p][ci].
    //      x is streamed exactly once -> non-temporal loads keep it out of the caches. ----
    const int l0 = lb * TILE_L;
    const float* xg = x + ((size_t)n * CIN + (size_t)g * CIG) * LEN;
    {
        const int NQ = XS_ROWS / 4;   // 66 quads per channel row
        for (int i = tid; i < CIG * NQ; i += 256) {
            int ci = i / NQ;
            int q  = i - ci * NQ;
            int gp = l0 - PADW + q * 4;           // multiple of 4
            v4f v = (v4f)(0.f);
            if (gp >= 0 && gp < LEN)              // aligned quad: fully valid or fully zero
                v = __builtin_nontemporal_load(
                        (const v4f*)(xg + (size_t)ci * LEN + gp));
            __bf16* d = Xs + (q * 4) * XS_P + ci;
            d[0 * XS_P] = (__bf16)v.x;
            d[1 * XS_P] = (__bf16)v.y;
            d[2 * XS_P] = (__bf16)v.z;
            d[3 * XS_P] = (__bf16)v.w;
        }
    }

    // ---- stage all 64 co weights once, K-permuted: Ws[co_r][k*64+ci].
    //      Weights ARE L2-reused across 512 blocks -> regular temporal loads. ----
    {
        const float* wg = w + (size_t)(g * COG) * KKTOT;
        for (int i = tid; i < COG * (KKTOT / 4); i += 256) {   // 64 * 144
            int cr = i / (KKTOT / 4);
            int s4 = (i - cr * (KKTOT / 4)) * 4;               // contiguous src = ci*9+k
            v4f v = *(const v4f*)(wg + (size_t)cr * KKTOT + s4);
#pragma unroll
            for (int j = 0; j < 4; ++j) {
                int s  = s4 + j;
                int ci = s / KW;
                int k  = s - ci * KW;
                Ws[cr * WS_P + k * CIG + ci] = (__bf16)v[j];
            }
        }
    }
    __syncthreads();

    // lane-constant fragment base pointers
    const __bf16* wrow  = Ws + lm * WS_P + 16 * half;              // A: M=lm, K contiguous by half
    const __bf16* xcol0 = Xs + (wv * 32 + lm) * XS_P + 16 * half;  // B tile 0: cols lm
    const __bf16* xcol1 = xcol0 + 16 * XS_P;                       // B tile 1: cols lm+16

    v8f acc[8] = {v8f{}, v8f{}, v8f{}, v8f{}, v8f{}, v8f{}, v8f{}, v8f{}};

#pragma unroll
    for (int t = 0; t < 18; ++t) {          // 576 / 32 K-chunks
        const int k   = t >> 1;             // conv tap (fixed within chunk)
        const int ci0 = (t & 1) << 5;       // 0 or 32
        const int kk0 = t << 5;
        const int xoff = k * XS_P + ci0;

        // Two B fragments (two column tiles), each reused by all 4 co-tiles
        v8bf b0lo = *(const v8bf*)(xcol0 + xoff);
        v8bf b0hi = *(const v8bf*)(xcol0 + xoff + 8);
        v8bf b1lo = *(const v8bf*)(xcol1 + xoff);
        v8bf b1hi = *(const v8bf*)(xcol1 + xoff + 8);
        v16bf b0, b1;
#pragma unroll
        for (int j = 0; j < 8; ++j) {
            b0[j] = b0lo[j]; b0[j + 8] = b0hi[j];
            b1[j] = b1lo[j]; b1[j + 8] = b1hi[j];
        }

#pragma unroll
        for (int ct = 0; ct < 4; ++ct) {
            const __bf16* wr = wrow + (size_t)(ct * 16) * WS_P + kk0;
            v8bf alo = *(const v8bf*)(wr);          // K_local = 16h+0..7
            v8bf ahi = *(const v8bf*)(wr + 8);      // K_local = 16h+8..15
            v16bf a;
#pragma unroll
            for (int j = 0; j < 8; ++j) { a[j] = alo[j]; a[j + 8] = ahi[j]; }
            // A held in registers feeds both column tiles (distinct accumulators: no hazards)
            acc[2 * ct + 0] = __builtin_amdgcn_wmma_f32_16x16x32_bf16(
                false, a, false, b0, (short)0, acc[2 * ct + 0], false, false);
            acc[2 * ct + 1] = __builtin_amdgcn_wmma_f32_16x16x32_bf16(
                false, a, false, b1, (short)0, acc[2 * ct + 1], false, false);
        }
    }

    // ---- epilogue: D VGPR r -> co = g*64 + ct*16 + r + 8*half; cols lm / lm+16.
    //      Output is write-once-never-read -> non-temporal stores. ----
    const size_t out_n = (size_t)n * COUT * LEN;
    const int lpos = l0 + wv * 32 + lm;
#pragma unroll
    for (int ct = 0; ct < 4; ++ct) {
#pragma unroll
        for (int r = 0; r < 8; ++r) {
            int co = g * COG + ct * 16 + r + 8 * half;
            float cx = ctx[n * COUT + co];
            float* op = out + out_n + (size_t)co * LEN + lpos;
            __builtin_nontemporal_store(acc[2 * ct + 0][r] + cx, op);
            __builtin_nontemporal_store(acc[2 * ct + 1][r] + cx, op + 16);
        }
    }
}

extern "C" void kernel_launch(void* const* d_in, const int* in_sizes, int n_in,
                              void* d_out, int out_size, void* d_ws, size_t ws_size,
                              hipStream_t stream) {
    const float* x    = (const float*)d_in[0];
    const float* c    = (const float*)d_in[1];
    const float* w    = (const float*)d_in[2];
    const float* cw   = (const float*)d_in[3];
    const float* bias = (const float*)d_in[4];
    float* out = (float*)d_out;
    float* ctx = (float*)d_ws;  // 32*256 floats of scratch

    ctx_bias_kernel<<<dim3(N_), dim3(COUT), 0, stream>>>(c, cw, bias, ctx);
    conv_wmma_kernel<<<dim3(N_ * 4 * (LEN / TILE_L)), dim3(256), 0, stream>>>(x, w, ctx, out);
    (void)in_sizes; (void)n_in; (void)out_size; (void)ws_size;
}
